// FP8Linear_75565654605930
// MI455X (gfx1250) — 8-bit matrix operand order repaired and checked by reading, compiled, not run on hardware
//
#include <hip/hip_runtime.h>
#include <stdint.h>

typedef int v8i __attribute__((ext_vector_type(8)));

#define B_  4
#define S_  2048
#define M_  8192
#define K_  4096
#define N_  4096

#define BM  128
#define BN  128
#define BK  128
#define LDA 144
#define LDB 144

__global__ void quantize_kernel(const float* __restrict__ x,
                                const float* __restrict__ input_scale,
                                int8_t* __restrict__ q) {
    const long idx = (long)blockIdx.x * blockDim.x + threadIdx.x;
    const long e   = idx * 4;
    const int  row = (int)(e >> 12);
    const float inv = 1.0f / input_scale[row & (S_ - 1)];
    const float4 v = *(const float4*)(x + e);
    int q0 = (int)rintf(v.x * inv); q0 = min(127, max(-128, q0));
    int q1 = (int)rintf(v.y * inv); q1 = min(127, max(-128, q1));
    int q2 = (int)rintf(v.z * inv); q2 = min(127, max(-128, q2));
    int q3 = (int)rintf(v.w * inv); q3 = min(127, max(-128, q3));
    const unsigned packed = (unsigned)(q0 & 255)        |
                            ((unsigned)(q1 & 255) << 8) |
                            ((unsigned)(q2 & 255) << 16)|
                            ((unsigned)(q3 & 255) << 24);
    *(unsigned*)(q + e) = packed;
}

__global__ void transpose_kernel(const int8_t* __restrict__ w,
                                 int8_t* __restrict__ wt) {
    __shared__ char tile[64][68];
    const int tc0 = blockIdx.x * 64;
    const int tr0 = blockIdx.y * 64;
    const int t = threadIdx.x;
    {
        const int r = t >> 2, c = (t & 3) * 16;
        *(int4*)&tile[r][c] =
            *(const int4*)(w + (long)(tr0 + r) * N_ + tc0 + c);
    }
    __syncthreads();
    {
        const int n = t >> 2, kc = (t & 3) * 16;
        union { char b[16]; int4 v; } u;
#pragma unroll
        for (int i = 0; i < 16; ++i) u.b[i] = tile[kc + i][n];
        *(int4*)(wt + (long)(tc0 + n) * K_ + tr0 + kc) = u.v;
    }
}

__global__ __launch_bounds__(256, 2)
void gemm_i8_kernel(const int8_t* __restrict__ qa,
                    const int8_t* __restrict__ wt,
                    const float* __restrict__ input_scale,
                    const float* __restrict__ weight_scale,
                    const float* __restrict__ bias,
                    float* __restrict__ out) {
    __shared__ char ldsA[2][BM * LDA];
    __shared__ char ldsB[2][BN * LDB];

    const int tid  = threadIdx.x;
    const int lane = tid & 31;
    const int wid  = tid >> 5;
    const int wm   = wid >> 2;
    const int wn   = wid & 3;
    const int lane15   = lane & 15;
    const int laneHalf = lane >> 4;

    const int m0 = blockIdx.y * BM;
    const int n0 = blockIdx.x * BN;

    const int lr = tid >> 1;
    const int lc = (tid & 1) * 64;
    const int8_t* gA = qa + (long)(m0 + lr) * K_ + lc;
    const int8_t* gB = wt + (long)(n0 + lr) * K_ + lc;

    const unsigned aDst0 = (unsigned)(unsigned long long)&ldsA[0][lr * LDA + lc];
    const unsigned bDst0 = (unsigned)(unsigned long long)&ldsB[0][lr * LDB + lc];
    const unsigned aBufStride = BM * LDA;
    const unsigned bBufStride = BN * LDB;

    v8i acc[4][2];
#pragma unroll
    for (int i = 0; i < 4; ++i)
#pragma unroll
        for (int j = 0; j < 2; ++j) {
            v8i z = {0, 0, 0, 0, 0, 0, 0, 0};
            acc[i][j] = z;
        }

    const int NK = K_ / BK;

#define ISSUE_STAGE(s)                                                         \
    do {                                                                       \
        const int8_t* pa_ = gA + (long)(s) * BK;                               \
        const int8_t* pb_ = gB + (long)(s) * BK;                               \
        const unsigned ad_ = aDst0 + ((s) & 1) * aBufStride;                   \
        const unsigned bd_ = bDst0 + ((s) & 1) * bBufStride;                   \
        asm volatile("global_load_async_to_lds_b128 %0, %1, off"               \
                     :: "v"(ad_), "v"(pa_) : "memory");                        \
        asm volatile("global_load_async_to_lds_b128 %0, %1, off offset:16"     \
                     :: "v"(ad_), "v"(pa_) : "memory");                        \
        asm volatile("global_load_async_to_lds_b128 %0, %1, off offset:32"     \
                     :: "v"(ad_), "v"(pa_) : "memory");                        \
        asm volatile("global_load_async_to_lds_b128 %0, %1, off offset:48"     \
                     :: "v"(ad_), "v"(pa_) : "memory");                        \
        asm volatile("global_load_async_to_lds_b128 %0, %1, off"               \
                     :: "v"(bd_), "v"(pb_) : "memory");                        \
        asm volatile("global_load_async_to_lds_b128 %0, %1, off offset:16"     \
                     :: "v"(bd_), "v"(pb_) : "memory");                        \
        asm volatile("global_load_async_to_lds_b128 %0, %1, off offset:32"     \
                     :: "v"(bd_), "v"(pb_) : "memory");                        \
        asm volatile("global_load_async_to_lds_b128 %0, %1, off offset:48"     \
                     :: "v"(bd_), "v"(pb_) : "memory");                        \
    } while (0)

    ISSUE_STAGE(0);
    ISSUE_STAGE(1);

    for (int kt = 0; kt < NK; ++kt) {
        const int buf = kt & 1;

        if (kt + 1 < NK) {
            asm volatile("s_wait_asynccnt 0x8" ::: "memory");
        } else {
            asm volatile("s_wait_asynccnt 0x0" ::: "memory");
        }
        __syncthreads();

#pragma unroll
        for (int kk = 0; kk < BK; kk += 64) {
            v8i afr[4];
#pragma unroll
            for (int mt = 0; mt < 4; ++mt) {
                const char* base = &ldsA[buf][(wm * 64 + mt * 16 + lane15) * LDA
                                              + kk + laneHalf * 16];
                const int2 x0 = *(const int2*)(base + 0);
                const int2 x1 = *(const int2*)(base + 8);
                const int2 x2 = *(const int2*)(base + 32);
                const int2 x3 = *(const int2*)(base + 40);
                v8i a;
                a[0] = x0.x; a[1] = x0.y; a[2] = x1.x; a[3] = x1.y;
                a[4] = x2.x; a[5] = x2.y; a[6] = x3.x; a[7] = x3.y;
                afr[mt] = a;
            }
            v8i bfr[2];
#pragma unroll
            for (int nt = 0; nt < 2; ++nt) {
                const char* base = &ldsB[buf][(wn * 32 + nt * 16 + lane15) * LDB
                                              + kk + laneHalf * 16];
                const int4 y0 = *(const int4*)(base + 0);
                const int4 y1 = *(const int4*)(base + 32);
                v8i b;
                b[0] = y0.x; b[1] = y0.y; b[2] = y0.z; b[3] = y0.w;
                b[4] = y1.x; b[5] = y1.y; b[6] = y1.z; b[7] = y1.w;
                bfr[nt] = b;
            }

#pragma unroll
            for (int mt = 0; mt < 4; ++mt)
#pragma unroll
                for (int nt = 0; nt < 2; ++nt)
                    acc[mt][nt] = __builtin_amdgcn_wmma_i32_16x16x64_iu8(
                         true, afr[mt],  true, bfr[nt],
                        acc[mt][nt],  false,  false);
        }

        __syncthreads();

        if (kt + 2 < NK) ISSUE_STAGE(kt + 2);
    }
#undef ISSUE_STAGE

#pragma unroll
    for (int nt = 0; nt < 2; ++nt) {
        const int col   = n0 + wn * 32 + nt * 16 + lane15;
        const float wsc = weight_scale[col];
        const float bb  = bias[col];
#pragma unroll
        for (int mt = 0; mt < 4; ++mt) {
            const int rbase = m0 + wm * 64 + mt * 16 + laneHalf * 8;
#pragma unroll
            for (int j = 0; j < 8; ++j) {
                const int r = rbase + j;
                const float asc = input_scale[r & (S_ - 1)];
                out[(long)r * N_ + col] =
                    (float)acc[mt][nt][j] * asc * wsc + bb;
            }
        }
    }
}

extern "C" void kernel_launch(void* const* d_in, const int* in_sizes, int n_in,
                              void* d_out, int out_size, void* d_ws, size_t ws_size,
                              hipStream_t stream) {
    const float*  input  = (const float*)d_in[0];
    const int8_t* weight = (const int8_t*)d_in[1];
    const float*  wscale = (const float*)d_in[2];
    const float*  iscale = (const float*)d_in[3];
    const float*  bias   = (const float*)d_in[4];
    float* out = (float*)d_out;

    int8_t* qa = (int8_t*)d_ws;
    int8_t* wt = qa + (size_t)M_ * K_;

    quantize_kernel<<<(M_ * (long)K_ / 4) / 256, 256, 0, stream>>>(input, iscale, qa);

    transpose_kernel<<<dim3(N_ / 64, K_ / 64), 256, 0, stream>>>(weight, wt);

    gemm_i8_kernel<<<dim3(N_ / BN, M_ / BM), 256, 0, stream>>>(
        qa, wt, iscale, wscale, bias, out);
}
